// Mel2LPC_81973745811483
// MI455X (gfx1250) — matrix operand order repaired and checked by reading, not compiled, not run on hardware
//
#include <hip/hip_runtime.h>
#include <hip/hip_bf16.h>
#include <math.h>

typedef __attribute__((ext_vector_type(16))) _Float16 v16h;
typedef __attribute__((ext_vector_type(8)))  float    v8f;
typedef __attribute__((ext_vector_type(4)))  unsigned int v4u;

#define N_FREQ      1025      // n_fft/2 + 1
#define N_FREQ_PAD  1040      // 65 tiles of 16
#define N_MELS      128
#define T_FRAMES    16384
#define REPEAT      512
#define TT_PER_WG   128       // 8 waves * 16 t-columns
#define LDSB_STRIDE 136       // halves: 128 + 8 pad (272B rows -> 4-bank shift/row)

union HV { v16h h; v4u u[2]; };

// Pre-kernel: inv_mel_basis f32[1025][128] -> f16[1040][128] (rows >=1025 zeroed).
// 266 KB result lives in L2 (192 MB) for the main kernel's 128 workgroups.
__global__ __launch_bounds__(256) void cvtA_kernel(const float* __restrict__ A,
                                                   _Float16* __restrict__ Ah) {
    int idx = blockIdx.x * 256 + threadIdx.x;
    if (idx >= N_FREQ_PAD * N_MELS) return;
    int f = idx / N_MELS;
    float v = (f < N_FREQ) ? A[idx] : 0.0f;
    Ah[idx] = (_Float16)v;
}

__global__ __launch_bounds__(256) void mel2lpc_kernel(
    const float*    __restrict__ mel,   // [128][16384]
    const _Float16* __restrict__ Ah,    // [1040][128] f16
    const float*    __restrict__ lagw,  // [5] lag window
    float*          __restrict__ out)   // [4][16384*512]
{
    __shared__ __align__(16) _Float16 ldsB[TT_PER_WG * LDSB_STRIDE];
    const int tid   = threadIdx.x;
    const int tbase = blockIdx.x * TT_PER_WG;

    // Stage exp(mel) as f16 into LDS, [t_local][k] layout (k contiguous).
    // Consecutive tid -> consecutive t: coalesced global reads.
    for (int idx = tid; idx < TT_PER_WG * (N_MELS / 2); idx += 256) {
        int t  = idx & (TT_PER_WG - 1);
        int kp = idx / TT_PER_WG;            // 0..63
        int k  = kp * 2;
        float a0 = __expf(mel[(size_t)k       * T_FRAMES + tbase + t]);
        float a1 = __expf(mel[(size_t)(k + 1) * T_FRAMES + tbase + t]);
        union { _Float16 h[2]; unsigned int u; } p;
        p.h[0] = (_Float16)a0;
        p.h[1] = (_Float16)a1;
        *(unsigned int*)&ldsB[t * LDSB_STRIDE + k] = p.u;
    }
    __syncthreads();

    const int lane = tid & 31;
    const int wv   = tid >> 5;       // wave id 0..7
    const int hi16 = lane >> 4;      // lane half (0/1)
    const int l16  = lane & 15;
    const int tloc = wv * 16 + l16;  // this lane's local t column

    const _Float16* bRow = &ldsB[tloc * LDSB_STRIDE];

    float lag[5] = {0.f, 0.f, 0.f, 0.f, 0.f};
    const float ang = 3.14159265358979323846f / 1024.0f;   // 2*pi / 2048

    for (int ff = 0; ff < N_FREQ_PAD / 16; ++ff) {
        const _Float16* aRow = Ah + (size_t)(ff * 16 + l16) * N_MELS;
        v8f c = {};
        #pragma unroll
        for (int kk = 0; kk < 4; ++kk) {
            HV a, b;
            // A (16x32 f16): lane half picks K off = 0 or 8; chunks at +0 and +16.
            int aoff = kk * 32 + hi16 * 8;
            a.u[0] = *(const v4u*)(aRow + aoff);
            a.u[1] = *(const v4u*)(aRow + aoff + 16);
            // B (32x16 f16): lane half picks K off = 0 or 8; chunks at +0 and +16 (matches A).
            int boff = kk * 32 + hi16 * 8;
            b.u[0] = *(const v4u*)(bRow + boff);
            b.u[1] = *(const v4u*)(bRow + boff + 16);
            c = __builtin_amdgcn_wmma_f32_16x16x32_f16(
                    false, a.h, false, b.h, (short)0, c, false, false);
        }
        // C layout: VGPR v, lane -> M = v + 8*hi16, N = l16.
        // Fold clamp/square + symmetric-IFFT cosine weights via Chebyshev recurrence.
        #pragma unroll
        for (int v = 0; v < 8; ++v) {
            int f = ff * 16 + v + hi16 * 8;
            if (f < N_FREQ) {
                float lin = fmaxf(c[v], 1e-12f);
                float p   = lin * lin;
                p *= ((f == 0) || (f == 1024)) ? (1.0f / 2048.0f) : (2.0f / 2048.0f);
                float c1 = __cosf(ang * (float)f);
                float c2 = 2.f * c1 * c1 - 1.f;
                float c3 = 2.f * c1 * c2 - c1;
                float c4 = 2.f * c1 * c3 - c2;
                lag[0] += p;
                lag[1] += p * c1;
                lag[2] += p * c2;
                lag[3] += p * c3;
                lag[4] += p * c4;
            }
        }
    }

    // Reduce lane halves (M=v vs M=v+8 rows), apply lag window.
    #pragma unroll
    for (int k = 0; k < 5; ++k) {
        lag[k] += __shfl_xor(lag[k], 16, 32);
        lag[k] *= lagw[k];
    }

    // Order-4 Levinson-Durbin (exact transcription of reference).
    float E  = lag[0];
    float acc = lag[1];
    float k0 = acc / E;
    E *= fmaxf(1.f - k0 * k0, 1e-5f);
    float lp0 = -k0;

    acc = lag[2] + lp0 * lag[1];
    float k1 = acc / E;
    E *= fmaxf(1.f - k1 * k1, 1e-5f);
    float n0 = lp0 - k1 * lp0;
    float lp1 = -k1;
    lp0 = n0;

    acc = lag[3] + lp0 * lag[2] + lp1 * lag[1];
    float k2 = acc / E;
    E *= fmaxf(1.f - k2 * k2, 1e-5f);
    n0 = lp0 - k2 * lp1;
    float n1 = lp1 - k2 * lp0;
    float lp2 = -k2;
    lp0 = n0; lp1 = n1;

    acc = lag[4] + lp0 * lag[3] + lp1 * lag[2] + lp2 * lag[1];
    float k3 = acc / E;
    n0 = lp0 - k3 * lp2;
    n1 = lp1 - k3 * lp1;
    float n2 = lp2 - k3 * lp0;
    float n3 = -k3;

    // out[l] = -lp[3-l], each value repeated 512x along t.
    float o[4] = { -n3, -n2, -n1, -n0 };

    // Wave-coalesced constant fills: each (t,l) run = 512 f32 = 2 KB,
    // written as 32 lanes x 4 float4 (512B per store instruction).
    const size_t rowStride = (size_t)T_FRAMES * REPEAT;
    for (int t = 0; t < 16; ++t) {
        size_t tglob = (size_t)tbase + wv * 16 + t;
        #pragma unroll
        for (int l = 0; l < 4; ++l) {
            float val = __shfl(o[l], t, 32);
            float4 v4 = make_float4(val, val, val, val);
            float* base = out + (size_t)l * rowStride + tglob * REPEAT;
            #pragma unroll
            for (int j = 0; j < 4; ++j)
                *(float4*)(base + lane * 4 + j * 128) = v4;
        }
    }
}

extern "C" void kernel_launch(void* const* d_in, const int* in_sizes, int n_in,
                              void* d_out, int out_size, void* d_ws, size_t ws_size,
                              hipStream_t stream) {
    const float* mel  = (const float*)d_in[0];   // (1,128,16384) f32
    const float* imb  = (const float*)d_in[1];   // (1025,128) f32
    const float* lagw = (const float*)d_in[2];   // (5,1) f32
    float* out = (float*)d_out;                  // (1,4,16384*512) f32
    _Float16* Ah = (_Float16*)d_ws;              // 1040*128 f16 = 266,240 B

    cvtA_kernel<<<(N_FREQ_PAD * N_MELS + 255) / 256, 256, 0, stream>>>(imb, Ah);
    mel2lpc_kernel<<<T_FRAMES / TT_PER_WG, 256, 0, stream>>>(mel, Ah, lagw, out);
}
